// MultiHeadAttention_62156766708193
// MI455X (gfx1250) — matrix operand order repaired and checked by reading, not compiled, not run on hardware
//
#include <hip/hip_runtime.h>
#include <cstdint>
#include <cstddef>

// ---------------- problem constants ----------------
constexpr int Bn   = 4;
constexpr int Sn   = 2048;
constexpr int Dn   = 1024;
constexpr int Hn   = 16;
constexpr int HDn  = 64;
constexpr int RK   = 16;
constexpr int Mn   = Bn * Sn;          // 8192 rows
constexpr float LORA_SCALE = 2.0f;     // 32/16

// ---------------- vector types ----------------
typedef __attribute__((ext_vector_type(16))) __bf16          v16bf;
typedef __attribute__((ext_vector_type(8)))  float           v8f;
typedef __attribute__((ext_vector_type(16))) unsigned short  v16u;
typedef __attribute__((ext_vector_type(8)))  unsigned short  v8u;
typedef __attribute__((ext_vector_type(4)))  unsigned int    v4u32;
typedef __attribute__((ext_vector_type(8)))  int             v8i32;
typedef __attribute__((ext_vector_type(4)))  int             v4i32;

__device__ __forceinline__ unsigned short f2bf(float f) {
  unsigned u = __builtin_bit_cast(unsigned, f);
  unsigned r = u + 0x7FFFu + ((u >> 16) & 1u);   // round-to-nearest-even
  return (unsigned short)(r >> 16);
}

// ---------------- Tensor Data Mover: 2D tile (64 rows x 128 bytes) global -> LDS ----------------
// D# built per CDNA5 ISA (data_size=4B, 32 dwords/row).
__device__ __forceinline__ void tdm_load_tile(const void* gsrc, unsigned lds_off,
                                              unsigned row_stride_bytes) {
  const unsigned long long ga = (unsigned long long)(size_t)gsrc;
  v4u32 g0;
  g0[0] = 1u;                                            // count=1 (valid), no gather
  g0[1] = lds_off;                                       // lds_addr (bytes)
  g0[2] = (unsigned)(ga & 0xFFFFFFFFu);                  // global_addr[31:0]
  g0[3] = (unsigned)((ga >> 32) & 0x01FFFFFFu)           // global_addr[56:32]
          | (2u << 30);                                  // type = 2 ("image")
  v8i32 g1;
  g1[0] = (int)(2u << 16);                               // wg_mask=0, data_size=2 (4B)
  g1[1] = (int)(32u << 16);                              // tensor_dim0[15:0] = 32
  g1[2] = (int)(64u << 16);                              // tensor_dim0 hi=0, tensor_dim1[15:0]=64
  g1[3] = (int)(32u << 16);                              // tensor_dim1 hi=0, tile_dim0=32
  g1[4] = (int)64u;                                      // tile_dim1=64, tile_dim2=0
  g1[5] = (int)(row_stride_bytes >> 2);                  // tensor_dim0_stride (dwords)
  g1[6] = 0;                                             // stride hi / dim1_stride lo
  g1[7] = 0;
  v4i32 z4; z4[0] = 0; z4[1] = 0; z4[2] = 0; z4[3] = 0;  // groups 2/3: 2D tensor
  v8i32 z8;
#pragma unroll
  for (int i = 0; i < 8; ++i) z8[i] = 0;
  __builtin_amdgcn_tensor_load_to_lds(g0, g1, z4, z4, z8, 0);
}

// ---------------- fp32 -> bf16 convert ----------------
__global__ void k_cvt_bf16(const float* __restrict__ src,
                           unsigned short* __restrict__ dst, int n) {
  int i = blockIdx.x * blockDim.x + threadIdx.x;
  if (i < n) dst[i] = f2bf(src[i]);
}

// ---------------- LoRA down projection: xa[m,r] = dot(x[m,:], A[r,:]) ----------------
__global__ void k_lora_down(const float* __restrict__ x,
                            const float* __restrict__ A,
                            float* __restrict__ xa) {
  int t = threadIdx.x;                       // 256 threads
  int m = blockIdx.x * 16 + (t >> 4);
  int r = t & 15;
  const float4* xr = (const float4*)(x + (size_t)m * Dn);
  const float4* ar = (const float4*)(A + (size_t)r * Dn);
  float acc = 0.f;
  for (int i = 0; i < Dn / 4; ++i) {
    float4 a = xr[i], b = ar[i];
    acc += a.x * b.x + a.y * b.y + a.z * b.z + a.w * b.w;
  }
  xa[(size_t)m * RK + r] = acc;
}

// ---------------- tiled WMMA GEMM: C = A(bf16)[M,K] * W(bf16)[N,K]^T + bias (+ LoRA) ----------------
// block: 256 threads (8 waves). Block tile 128(M) x 64(N); wave tile 16 x 64.
template <bool HAS_LORA, bool OUT_F32>
__global__ void __launch_bounds__(256)
k_gemm(const unsigned short* __restrict__ Abf,
       const unsigned short* __restrict__ Wbf,
       const float* __restrict__ bias,
       const float* __restrict__ xa,     // [M,16]  (LoRA)
       const float* __restrict__ Bm,     // [N,16]  (LoRA up-matrix, row-major)
       unsigned short* __restrict__ outb,
       float* __restrict__ outf,
       int Kn, int Nn) {
  const int lane = threadIdx.x & 31;
  const int wave = threadIdx.x >> 5;
  const int half = lane >> 4;
  const int l16  = lane & 15;
  const int mrow = blockIdx.x * 128 + wave * 16;
  const int col0 = blockIdx.y * 64;

  v8f acc[4];
#pragma unroll
  for (int f = 0; f < 4; ++f)
#pragma unroll
    for (int i = 0; i < 8; ++i) acc[f][i] = 0.f;

  const int akb = half * 16;                     // A-fragment K base per half-wave
  for (int k0 = 0; k0 < Kn; k0 += 32) {
    // A fragment: lane = row (m), half-wave holds K 0..15 / 16..31 (two 8-elem chunks)
    const unsigned short* ap = Abf + (size_t)(mrow + l16) * Kn + k0 + akb;
    v8u a0 = *(const v8u*)(ap);
    v8u a1 = *(const v8u*)(ap + 8);
    v16u au;
#pragma unroll
    for (int i = 0; i < 8; ++i) { au[i] = a0[i]; au[i + 8] = a1[i]; }
    v16bf af = __builtin_bit_cast(v16bf, au);

#pragma unroll
    for (int f = 0; f < 4; ++f) {
      // B fragment: lane = column (n), half-wave holds K 0..15 / 16..31 -> one 32B read
      const unsigned short* bp =
          Wbf + (size_t)(col0 + f * 16 + l16) * Kn + k0 + half * 16;
      v16u bu = *(const v16u*)bp;
      v16bf bfrag = __builtin_bit_cast(v16bf, bu);
      acc[f] = __builtin_amdgcn_wmma_f32_16x16x32_bf16(
          false, af, false, bfrag, (short)0, acc[f], false, false);
    }
  }

  // epilogue: bias + optional rank-16 LoRA, store
#pragma unroll
  for (int f = 0; f < 4; ++f) {
    const int n = col0 + f * 16 + l16;
    const float bv = bias[n];
    float lf[RK];
    if (HAS_LORA) {
#pragma unroll
      for (int j = 0; j < RK; ++j) lf[j] = Bm[(size_t)n * RK + j];
    }
#pragma unroll
    for (int r = 0; r < 8; ++r) {
      const int m = mrow + r + half * 8;
      float c = acc[f][r] + bv;
      if (HAS_LORA) {
        const float* xr = xa + (size_t)m * RK;
        float s = 0.f;
#pragma unroll
        for (int j = 0; j < RK; ++j) s += xr[j] * lf[j];
        c += LORA_SCALE * s;
      }
      if (OUT_F32) outf[(size_t)m * Nn + n] = c;
      else         outb[(size_t)m * Nn + n] = f2bf(c);
    }
  }
}

// ---------------- flash attention: per (q-tile 64, head, batch) ----------------
// block: 128 threads = 4 waves, each wave owns 16 query rows.
// K/V tiles staged by the Tensor Data Mover (TENSORcnt), WMMA for both GEMMs.
__global__ void __launch_bounds__(128)
k_attn(const unsigned short* __restrict__ q,
       const unsigned short* __restrict__ k,
       const unsigned short* __restrict__ v,
       const float* __restrict__ mask,          // [B,1,1,S] -> [B*S]
       unsigned short* __restrict__ ctx) {
  __shared__ alignas(64) unsigned short Ks[64 * 64];
  __shared__ alignas(64) unsigned short Vs[64 * 64];
  __shared__ alignas(64) unsigned short Ps[4][16 * 64];

  const int qt   = blockIdx.x;          // 0..31
  const int h    = blockIdx.y;          // 0..15
  const int b    = blockIdx.z;          // 0..3
  const int lane = threadIdx.x & 31;
  const int wave = threadIdx.x >> 5;
  const int half = lane >> 4;
  const int l16  = lane & 15;

  const size_t rowbase = (size_t)b * Sn;
  const int hd0 = h * HDn;
  const unsigned ldsK = (unsigned)(size_t)(void*)Ks;   // LDS byte offset (flat addr low bits)
  const unsigned ldsV = (unsigned)(size_t)(void*)Vs;

  // Q A-fragments (16 rows x 64 hd), two K-steps of 32
  v16bf qa[2];
  {
    const int m = qt * 64 + wave * 16 + l16;
    const unsigned short* qp = q + (rowbase + m) * Dn + hd0;
#pragma unroll
    for (int ks = 0; ks < 2; ++ks) {
      const int kb = ks * 32 + half * 16;
      v8u a0 = *(const v8u*)(qp + kb);
      v8u a1 = *(const v8u*)(qp + kb + 8);
      v16u au;
#pragma unroll
      for (int i = 0; i < 8; ++i) { au[i] = a0[i]; au[i + 8] = a1[i]; }
      qa[ks] = __builtin_bit_cast(v16bf, au);
    }
  }

  v8f o[4];
#pragma unroll
  for (int f = 0; f < 4; ++f)
#pragma unroll
    for (int i = 0; i < 8; ++i) o[f][i] = 0.f;
  float mrow[8], lrow[8];
#pragma unroll
  for (int r = 0; r < 8; ++r) { mrow[r] = -1e30f; lrow[r] = 0.f; }

  for (int kt = 0; kt < Sn / 64; ++kt) {
    // TDM staging of K/V tiles (64 keys x 64 bf16 = 64 x 128B), one wave issues the DMA
    if (wave == 0) {
      const unsigned short* kg = k + (rowbase + kt * 64) * Dn + hd0;
      const unsigned short* vg = v + (rowbase + kt * 64) * Dn + hd0;
      tdm_load_tile(kg, ldsK, Dn * 2);
      tdm_load_tile(vg, ldsV, Dn * 2);
      __builtin_amdgcn_s_wait_tensorcnt(0);
    }
    __syncthreads();

    // S = Q * K^T  (4 key sub-tiles of 16)
    v8f s[4];
#pragma unroll
    for (int f = 0; f < 4; ++f) {
      v8f c;
#pragma unroll
      for (int i = 0; i < 8; ++i) c[i] = 0.f;
#pragma unroll
      for (int ks = 0; ks < 2; ++ks) {
        const unsigned short* bp = Ks + (f * 16 + l16) * 64 + ks * 32 + half * 16;
        v16u bu = *(const v16u*)bp;
        v16bf bfrag = __builtin_bit_cast(v16bf, bu);
        c = __builtin_amdgcn_wmma_f32_16x16x32_bf16(
            false, qa[ks], false, bfrag, (short)0, c, false, false);
      }
      const float mv = mask[rowbase + kt * 64 + f * 16 + l16];
#pragma unroll
      for (int r = 0; r < 8; ++r) s[f][r] = c[r] * 0.125f + mv;
    }

    // online softmax: row max (rows live in half-waves -> 16-lane xor reduce)
    float corr[8];
#pragma unroll
    for (int r = 0; r < 8; ++r) {
      float tm = fmaxf(fmaxf(s[0][r], s[1][r]), fmaxf(s[2][r], s[3][r]));
      tm = fmaxf(tm, __shfl_xor(tm, 1, 32));
      tm = fmaxf(tm, __shfl_xor(tm, 2, 32));
      tm = fmaxf(tm, __shfl_xor(tm, 4, 32));
      tm = fmaxf(tm, __shfl_xor(tm, 8, 32));
      const float mnew = fmaxf(mrow[r], tm);
      corr[r] = __expf(mrow[r] - mnew);
      mrow[r] = mnew;
    }

    // P = exp(S - m): C-layout -> A-layout via wave-private LDS (DS ops in-order per wave)
    unsigned short* pp = Ps[wave];
    float rs[8];
#pragma unroll
    for (int r = 0; r < 8; ++r) rs[r] = 0.f;
#pragma unroll
    for (int f = 0; f < 4; ++f)
#pragma unroll
      for (int r = 0; r < 8; ++r) {
        const float p = __expf(s[f][r] - mrow[r]);
        rs[r] += p;
        pp[(r + half * 8) * 64 + f * 16 + l16] = f2bf(p);
      }
#pragma unroll
    for (int r = 0; r < 8; ++r) {
      float t = rs[r];
      t += __shfl_xor(t, 1, 32);
      t += __shfl_xor(t, 2, 32);
      t += __shfl_xor(t, 4, 32);
      t += __shfl_xor(t, 8, 32);
      lrow[r] = lrow[r] * corr[r] + t;
    }
#pragma unroll
    for (int f = 0; f < 4; ++f)
#pragma unroll
      for (int r = 0; r < 8; ++r) o[f][r] *= corr[r];

    // reload P as A-fragments
    v16bf pa[2];
#pragma unroll
    for (int ks = 0; ks < 2; ++ks) {
      const unsigned short* ap = pp + l16 * 64 + ks * 32 + half * 16;
      v8u a0 = *(const v8u*)ap;
      v8u a1 = *(const v8u*)(ap + 8);
      v16u au;
#pragma unroll
      for (int i = 0; i < 8; ++i) { au[i] = a0[i]; au[i + 8] = a1[i]; }
      pa[ks] = __builtin_bit_cast(v16bf, au);
    }

    // O += P * V   (B fragment = keys x hd column reads from LDS)
#pragma unroll
    for (int f = 0; f < 4; ++f)
#pragma unroll
      for (int ks = 0; ks < 2; ++ks) {
        v16u bu;
#pragma unroll
        for (int i = 0; i < 16; ++i)
          bu[i] = Vs[(ks * 32 + half * 16 + i) * 64 + f * 16 + l16];
        v16bf bfrag = __builtin_bit_cast(v16bf, bu);
        o[f] = __builtin_amdgcn_wmma_f32_16x16x32_bf16(
            false, pa[ks], false, bfrag, (short)0, o[f], false, false);
      }
    __syncthreads();
  }

  // normalize and store ctx (bf16, [B*S, D] with head-interleaved columns)
#pragma unroll
  for (int r = 0; r < 8; ++r) {
    const float inv = 1.0f / lrow[r];
    const int m = qt * 64 + wave * 16 + r + half * 8;
    unsigned short* cp = ctx + (rowbase + m) * Dn + hd0;
#pragma unroll
    for (int f = 0; f < 4; ++f) cp[f * 16 + l16] = f2bf(o[f][r] * inv);
  }
}

// ---------------- host-side launch ----------------
extern "C" void kernel_launch(void* const* d_in, const int* in_sizes, int n_in,
                              void* d_out, int out_size, void* d_ws, size_t ws_size,
                              hipStream_t stream) {
  const float* x    = (const float*)d_in[0];
  const float* mask = (const float*)d_in[1];
  const float* Wq   = (const float*)d_in[2];
  const float* bq   = (const float*)d_in[3];
  const float* Aq   = (const float*)d_in[4];
  const float* Bq   = (const float*)d_in[5];
  const float* Wk   = (const float*)d_in[6];
  const float* bk   = (const float*)d_in[7];
  const float* Wv   = (const float*)d_in[8];
  const float* bv   = (const float*)d_in[9];
  const float* Av   = (const float*)d_in[10];
  const float* Bv   = (const float*)d_in[11];
  const float* Wo   = (const float*)d_in[12];
  const float* bo   = (const float*)d_in[13];
  float* out = (float*)d_out;

  // workspace layout (bytes, all offsets 256-aligned)
  char* ws = (char*)d_ws;
  unsigned short* xb   = (unsigned short*)(ws);                       // 16 MB
  unsigned short* Wqb  = (unsigned short*)(ws + (16u << 20));         //  2 MB
  unsigned short* Wkb  = (unsigned short*)(ws + (18u << 20));
  unsigned short* Wvb  = (unsigned short*)(ws + (20u << 20));
  unsigned short* Wob  = (unsigned short*)(ws + (22u << 20));
  unsigned short* qb   = (unsigned short*)(ws + (24u << 20));         // 16 MB
  unsigned short* kb   = (unsigned short*)(ws + (40u << 20));
  unsigned short* vb   = (unsigned short*)(ws + (56u << 20));
  unsigned short* ctxb = (unsigned short*)(ws + (72u << 20));
  float*          xaq  = (float*)(ws + (88u << 20));                  // 512 KB
  float*          xav  = (float*)(ws + (88u << 20) + (1u << 19));

  const int nX = Mn * Dn;        // 8388608
  const int nW = Dn * Dn;        // 1048576

  k_cvt_bf16<<<nX / 256, 256, 0, stream>>>(x,  xb,  nX);
  k_cvt_bf16<<<nW / 256, 256, 0, stream>>>(Wq, Wqb, nW);
  k_cvt_bf16<<<nW / 256, 256, 0, stream>>>(Wk, Wkb, nW);
  k_cvt_bf16<<<nW / 256, 256, 0, stream>>>(Wv, Wvb, nW);
  k_cvt_bf16<<<nW / 256, 256, 0, stream>>>(Wo, Wob, nW);

  k_lora_down<<<Mn / 16, 256, 0, stream>>>(x, Aq, xaq);
  k_lora_down<<<Mn / 16, 256, 0, stream>>>(x, Av, xav);

  dim3 ggrid(Mn / 128, Dn / 64);   // 64 x 16
  k_gemm<true,  false><<<ggrid, 256, 0, stream>>>(xb, Wqb, bq, xaq, Bq, qb, nullptr, Dn, Dn);
  k_gemm<false, false><<<ggrid, 256, 0, stream>>>(xb, Wkb, bk, nullptr, nullptr, kb, nullptr, Dn, Dn);
  k_gemm<true,  false><<<ggrid, 256, 0, stream>>>(xb, Wvb, bv, xav, Bv, vb, nullptr, Dn, Dn);

  dim3 agrid(Sn / 64, Hn, Bn);     // 32 x 16 x 4
  k_attn<<<agrid, 128, 0, stream>>>(qb, kb, vb, mask, ctxb);

  k_gemm<false, true><<<ggrid, 256, 0, stream>>>(ctxb, Wob, bo, nullptr, nullptr, nullptr, out, Dn, Dn);
}
